// kernelLogLikelihood_62560493633740
// MI455X (gfx1250) — matrix operand order repaired and checked by reading, not compiled, not run on hardware
//
#include <hip/hip_runtime.h>
#include <hip/hip_bf16.h>
#include <math.h>

// ---------------------------------------------------------------------------
// Problem constants (from reference setup_inputs)
// ---------------------------------------------------------------------------
#define B_ROWS 4096     // batch
#define D_IN   512      // input dim
#define D_PCA  100      // pca dim (K of the big GEMM)
#define N_REF  50000    // reference points (divisible by 16, not by 32)
#define KP     128      // K padded to multiple of 32 for bf16 WMMA
#define NSPLIT 98       // N split into 98 chunks of CHUNK
#define CHUNK  512

typedef __attribute__((ext_vector_type(16))) __bf16 v16bf;
typedef __attribute__((ext_vector_type(8)))  __bf16 v8bf;
typedef __attribute__((ext_vector_type(8)))  float  v8f;

union BF16x16 { v16bf v; v8bf h[2]; };

__device__ __forceinline__ v8f wmma_bf16(v16bf a, v16bf b, v8f c) {
  // D = A(16x32 bf16) * B(32x16 bf16) + C(16x16 f32)
  return __builtin_amdgcn_wmma_f32_16x16x32_bf16(
      /*neg_a=*/false, a, /*neg_b=*/false, b,
      /*c_mod=*/(short)0, c, /*reuse_a=*/false, /*reuse_b=*/false);
}

// ---------------------------------------------------------------------------
// Kernel 1: h = x @ W^T + b   -> split into bf16 hi/lo, K-padded; plus ||h||^2
// ---------------------------------------------------------------------------
__global__ __launch_bounds__(128) void prep_h_kernel(
    const float* __restrict__ x, const float* __restrict__ W,
    const float* __restrict__ bpca,
    unsigned short* __restrict__ Hhi_u, unsigned short* __restrict__ Hlo_u,
    float* __restrict__ hsq)
{
  __shared__ float xrow[D_IN];
  __shared__ float red[128];
  const int row = blockIdx.x;
  const int t   = threadIdx.x;

  for (int i = t; i < D_IN; i += 128) xrow[i] = x[row * D_IN + i];
  __syncthreads();

  float v = 0.0f;
  if (t < D_PCA) {
    float acc = bpca[t];
    const float* w = W + t * D_IN;
    #pragma unroll 4
    for (int i = 0; i < D_IN; ++i) acc = fmaf(xrow[i], w[i], acc);
    v = acc;
  }

  const __bf16 hb = (__bf16)v;
  const __bf16 lb = (__bf16)(v - (float)hb);
  __bf16* Hh = (__bf16*)Hhi_u;
  __bf16* Hl = (__bf16*)Hlo_u;
  Hh[row * KP + t] = hb;   // threads t>=100 carry v==0 -> zero padding
  Hl[row * KP + t] = lb;

  red[t] = v * v;
  __syncthreads();
  #pragma unroll
  for (int s = 64; s > 0; s >>= 1) {
    if (t < s) red[t] += red[t + s];
    __syncthreads();
  }
  if (t == 0) hsq[row] = red[0];
}

// ---------------------------------------------------------------------------
// Kernel 2: X_ref -> bf16 hi/lo (K-padded) and  nxqs_j = -||X_j||^2 * s
// where s = log2(e)/(2*bw^2), so the GEMM inner loop needs one fma per logit.
// ---------------------------------------------------------------------------
__global__ __launch_bounds__(128) void prep_x_kernel(
    const float* __restrict__ Xr, const float* __restrict__ bwp,
    unsigned short* __restrict__ Xhi_u, unsigned short* __restrict__ Xlo_u,
    float* __restrict__ nXsqs)
{
  __shared__ float red[128];
  const int row = blockIdx.x;
  const int t   = threadIdx.x;

  const float v = (t < D_PCA) ? Xr[row * D_PCA + t] : 0.0f;
  const __bf16 hb = (__bf16)v;
  const __bf16 lb = (__bf16)(v - (float)hb);
  __bf16* Xh = (__bf16*)Xhi_u;
  __bf16* Xl = (__bf16*)Xlo_u;
  Xh[row * KP + t] = hb;
  Xl[row * KP + t] = lb;

  red[t] = v * v;
  __syncthreads();
  #pragma unroll
  for (int s = 64; s > 0; s >>= 1) {
    if (t < s) red[t] += red[t + s];
    __syncthreads();
  }
  if (t == 0) {
    const float bw = bwp[0];
    const float sc = 1.4426950408889634f / (2.0f * bw * bw);
    nXsqs[row] = -red[0] * sc;
  }
}

// ---------------------------------------------------------------------------
// Kernel 3: fused GEMM (bf16x3 split-precision WMMA) + online log-sum-exp.
// grid = (B_ROWS/128, NSPLIT); block = 256 threads = 8 waves.
// Each wave owns 16 batch rows; A fragments (hi+lo, 4 K-steps) stay in VGPRs.
// Per iteration: TWO 16-column tiles, each with TWO accumulators (hi*hi and
// cross terms) -> 24 WMMAs in 4 short dependency chains. B-side addressing
// uses lane-local pointers bumped by a constant stride so all 16 loads per
// iteration are immediate-offset global_load_b128.
// ---------------------------------------------------------------------------
__global__ __launch_bounds__(256) void gemm_lse_kernel(
    const unsigned short* __restrict__ Hhi_u, const unsigned short* __restrict__ Hlo_u,
    const unsigned short* __restrict__ Xhi_u, const unsigned short* __restrict__ Xlo_u,
    const float* __restrict__ nXsqs, const float* __restrict__ bwp,
    float* __restrict__ pmax, float* __restrict__ psum)
{
  const __bf16* Hhi = (const __bf16*)Hhi_u;
  const __bf16* Hlo = (const __bf16*)Hlo_u;

  const int wave = threadIdx.x >> 5;
  const int lane = threadIdx.x & 31;
  const int l16  = lane & 15;
  const int half = lane >> 4;
  const int mrow = blockIdx.x * 128 + wave * 16 + l16;

  const float bw    = bwp[0];
  const float two_s = 2.0f * 1.4426950408889634f / (2.0f * bw * bw);

  // ---- A fragments (resident for the whole N-chunk) ----
  v16bf ahi[4], alo[4];
  #pragma unroll
  for (int ks = 0; ks < 4; ++ks) {
    const int kk = ks * 32;
    BF16x16 a;
    a.h[0] = *(const v8bf*)(Hhi + mrow * KP + kk + 16 * half);
    a.h[1] = *(const v8bf*)(Hhi + mrow * KP + kk + 16 * half + 8);
    ahi[ks] = a.v;
    a.h[0] = *(const v8bf*)(Hlo + mrow * KP + kk + 16 * half);
    a.h[1] = *(const v8bf*)(Hlo + mrow * KP + kk + 16 * half + 8);
    alo[ks] = a.v;
  }

  float rmax[8], rsum[8];
  #pragma unroll
  for (int r = 0; r < 8; ++r) { rmax[r] = -__builtin_inff(); rsum[r] = 0.0f; }

  const int nbeg = blockIdx.y * CHUNK;
  const int nend = (nbeg + CHUNK < N_REF) ? (nbeg + CHUNK) : N_REF;
  const int span = nend - nbeg;

  // lane-local streaming pointers (constant immediate offsets inside the loop)
  const __bf16* pXhi = (const __bf16*)Xhi_u + (size_t)(nbeg + l16) * KP + 16 * half;
  const __bf16* pXlo = (const __bf16*)Xlo_u + (size_t)(nbeg + l16) * KP + 16 * half;
  const float*  pXq  = nXsqs + nbeg + l16;

  // ---- main loop: two 16-column tiles per iteration ----
  for (int it = span >> 5; it > 0; --it) {
    const float nxq0 = pXq[0];
    const float nxq1 = pXq[16];

    v8f hh0 = {}, xx0 = {}, hh1 = {}, xx1 = {};
    #pragma unroll
    for (int ks = 0; ks < 4; ++ks) {
      const v16bf bhi0 = *(const v16bf*)(pXhi + ks * 32);
      const v16bf blo0 = *(const v16bf*)(pXlo + ks * 32);
      const v16bf bhi1 = *(const v16bf*)(pXhi + 16 * KP + ks * 32);
      const v16bf blo1 = *(const v16bf*)(pXlo + 16 * KP + ks * 32);
      hh0 = wmma_bf16(ahi[ks], bhi0, hh0);   // tile0 hi*hi   (chain depth 4)
      xx0 = wmma_bf16(ahi[ks], blo0, xx0);   // tile0 hi*lo   (chain depth 8)
      xx0 = wmma_bf16(alo[ks], bhi0, xx0);   // tile0 lo*hi
      hh1 = wmma_bf16(ahi[ks], bhi1, hh1);   // tile1 hi*hi
      xx1 = wmma_bf16(ahi[ks], blo1, xx1);   // tile1 hi*lo
      xx1 = wmma_bf16(alo[ks], bhi1, xx1);   // tile1 lo*hi
    }
    pXhi += 32 * KP;
    pXlo += 32 * KP;
    pXq  += 32;

    #pragma unroll
    for (int r = 0; r < 8; ++r) {
      const float a0 = fmaf(hh0[r] + xx0[r], two_s, nxq0);
      const float a1 = fmaf(hh1[r] + xx1[r], two_s, nxq1);
      // single combined rescale: 3 exp2 per row instead of 4 (v_max3_f32)
      const float nm = fmaxf(fmaxf(rmax[r], a0), a1);
      rsum[r] = rsum[r] * exp2f(rmax[r] - nm) + (exp2f(a0 - nm) + exp2f(a1 - nm));
      rmax[r] = nm;
    }
  }

  // ---- 16-column tail (only the last chunk hits this; 50000 % 32 != 0) ----
  if (span & 16) {
    const float nxq0 = pXq[0];
    v8f hh0 = {}, xx0 = {};
    #pragma unroll
    for (int ks = 0; ks < 4; ++ks) {
      const v16bf bhi0 = *(const v16bf*)(pXhi + ks * 32);
      const v16bf blo0 = *(const v16bf*)(pXlo + ks * 32);
      hh0 = wmma_bf16(ahi[ks], bhi0, hh0);
      xx0 = wmma_bf16(ahi[ks], blo0, xx0);
      xx0 = wmma_bf16(alo[ks], bhi0, xx0);
    }
    #pragma unroll
    for (int r = 0; r < 8; ++r) {
      const float a0 = fmaf(hh0[r] + xx0[r], two_s, nxq0);
      const float nm = fmaxf(rmax[r], a0);
      rsum[r] = rsum[r] * exp2f(rmax[r] - nm) + exp2f(a0 - nm);
      rmax[r] = nm;
    }
  }

  // ---- reduce across the 16 columns held by the lanes of each half-wave ----
  #pragma unroll
  for (int mask = 1; mask < 16; mask <<= 1) {
    #pragma unroll
    for (int r = 0; r < 8; ++r) {
      const float om = __shfl_xor(rmax[r], mask, 32);
      const float os = __shfl_xor(rsum[r], mask, 32);
      const float nm = fmaxf(rmax[r], om);
      rsum[r] = rsum[r] * exp2f(rmax[r] - nm) + os * exp2f(om - nm);
      rmax[r] = nm;
    }
  }

  if (l16 == 0) {
    #pragma unroll
    for (int r = 0; r < 8; ++r) {
      const int row = blockIdx.x * 128 + wave * 16 + half * 8 + r;
      pmax[row * NSPLIT + blockIdx.y] = rmax[r];
      psum[row * NSPLIT + blockIdx.y] = rsum[r];
    }
  }
}

// ---------------------------------------------------------------------------
// Kernel 4: combine per-chunk partial (max, sum) -> final log-likelihood.
// ---------------------------------------------------------------------------
__global__ __launch_bounds__(256) void finalize_kernel(
    const float* __restrict__ pmax, const float* __restrict__ psum,
    const float* __restrict__ hsq, const float* __restrict__ bwp,
    float* __restrict__ out)
{
  const int row = blockIdx.x * blockDim.x + threadIdx.x;
  if (row >= B_ROWS) return;

  float M = -__builtin_inff();
  for (int ns = 0; ns < NSPLIT; ++ns) M = fmaxf(M, pmax[row * NSPLIT + ns]);
  float ssum = 0.0f;
  for (int ns = 0; ns < NSPLIT; ++ns)
    ssum += psum[row * NSPLIT + ns] * exp2f(pmax[row * NSPLIT + ns] - M);

  const float bw   = bwp[0];
  const float inv2 = 1.0f / (2.0f * bw * bw);
  // coeff = -ln(N) - ln(2*pi)*d/2 - ln(bw)*d   (ln(2*pi)/2 = 0.9189385332)
  const float coeff = -logf((float)N_REF)
                      - 0.91893853320467274f * (float)D_PCA
                      - logf(bw) * (float)D_PCA;
  // LSE_e = ln2 * (M + log2(sum)), then subtract the factored-out ||h||^2 term
  out[row] = 0.69314718055994531f * (M + log2f(ssum)) - hsq[row] * inv2 + coeff;
}

// ---------------------------------------------------------------------------
// Launch glue
// ---------------------------------------------------------------------------
extern "C" void kernel_launch(void* const* d_in, const int* in_sizes, int n_in,
                              void* d_out, int out_size, void* d_ws, size_t ws_size,
                              hipStream_t stream) {
  (void)in_sizes; (void)n_in; (void)out_size; (void)ws_size;

  const float* x    = (const float*)d_in[0];   // [4096,512]
  const float* W    = (const float*)d_in[1];   // [100,512]
  const float* bpca = (const float*)d_in[2];   // [100]
  const float* Xr   = (const float*)d_in[3];   // [50000,100]
  const float* bw   = (const float*)d_in[4];   // scalar
  float* out        = (float*)d_out;           // [4096]

  char* ws = (char*)d_ws;
  size_t off = 0;
  auto take = [&](size_t bytes) {
    char* p = ws + off;
    off = (off + bytes + 255) & ~(size_t)255;
    return p;
  };

  unsigned short* Hhi = (unsigned short*)take((size_t)B_ROWS * KP * 2);
  unsigned short* Hlo = (unsigned short*)take((size_t)B_ROWS * KP * 2);
  float*          hsq = (float*)        take((size_t)B_ROWS * 4);
  unsigned short* Xhi = (unsigned short*)take((size_t)N_REF * KP * 2);
  unsigned short* Xlo = (unsigned short*)take((size_t)N_REF * KP * 2);
  float*          nXq = (float*)        take((size_t)N_REF * 4);
  float*          pmx = (float*)        take((size_t)B_ROWS * NSPLIT * 4);
  float*          psm = (float*)        take((size_t)B_ROWS * NSPLIT * 4);

  prep_h_kernel<<<B_ROWS, 128, 0, stream>>>(x, W, bpca, Hhi, Hlo, hsq);
  prep_x_kernel<<<N_REF, 128, 0, stream>>>(Xr, bw, Xhi, Xlo, nXq);

  dim3 grid(B_ROWS / 128, NSPLIT);
  gemm_lse_kernel<<<grid, 256, 0, stream>>>(Hhi, Hlo, Xhi, Xlo, nXq, bw, pmx, psm);

  finalize_kernel<<<(B_ROWS + 255) / 256, 256, 0, stream>>>(pmx, psm, hsq, bw, out);
}
